// GCNSI_77567109366385
// MI455X (gfx1250) — compile-verified, matrix operand order checked by reading, not run on hardware
//
#include <hip/hip_runtime.h>

// ---------------- types ----------------
typedef __bf16 bf16;
typedef __attribute__((ext_vector_type(16))) __bf16 v16bf;
typedef __attribute__((ext_vector_type(8)))  __bf16 v8bf;
typedef __attribute__((ext_vector_type(8)))  float  v8f;

#define HID 256

__device__ __forceinline__ float bf2f(bf16 v) {
    unsigned short s = __builtin_bit_cast(unsigned short, v);
    unsigned u = ((unsigned)s) << 16;
    return __builtin_bit_cast(float, u);
}
__device__ __forceinline__ bf16 f2bf(float f) {
    unsigned u = __builtin_bit_cast(unsigned, f);
    u += 0x7FFFu + ((u >> 16) & 1u);           // round-to-nearest-even
    unsigned short s = (unsigned short)(u >> 16);
    return __builtin_bit_cast(bf16, s);
}

// ---------------- graph norm ----------------
__global__ __launch_bounds__(256) void k_zero_f(float* __restrict__ p, int n) {
    int i = blockIdx.x * blockDim.x + threadIdx.x;
    if (i < n) p[i] = 0.0f;
}

__global__ __launch_bounds__(256) void k_deg(const int* __restrict__ ei, float* __restrict__ deg, int nE) {
    int e = blockIdx.x * blockDim.x + threadIdx.x;
    if (e < nE) atomicAdd(&deg[ei[nE + e]], 1.0f);
}

__global__ __launch_bounds__(256) void k_dinv(float* __restrict__ deg, int n) {
    int i = blockIdx.x * blockDim.x + threadIdx.x;
    if (i < n) deg[i] = __frsqrt_rn(deg[i] + 1.0f);   // self-loop included
}

// ---------------- W pack: [K=256][N=256] f32 -> WMMA B fragments (bf16) ----------------
// Fragment f = (kb/32)*16 + (n0/16). Each of 32 lanes holds 16 contiguous bf16.
// Lane l: n = l&15, half = l>>4; element e holds W[kb + Ke][n0 + n],
// Ke = (e<8 ? e : e+8) + half*8   (per CDNA5 16-bit 32x16 fragment layout).
__global__ __launch_bounds__(128) void k_packw(const float* __restrict__ W, bf16* __restrict__ Wp) {
    int t = blockIdx.x * blockDim.x + threadIdx.x;     // 128 frags * 32 lanes = 4096
    if (t >= 128 * 32) return;
    int f    = t >> 5;
    int lane = t & 31;
    int kb = (f >> 4) * 32;
    int n0 = (f & 15) * 16;
    int n    = lane & 15;
    int half = lane >> 4;
    bf16* dst = Wp + (size_t)t * 16;
#pragma unroll
    for (int e = 0; e < 16; ++e) {
        int Ke = ((e < 8) ? e : e + 8) + half * 8;
        dst[e] = f2bf(W[(size_t)(kb + Ke) * HID + n0 + n]);
    }
}

// ---------------- layer 0 dense transform: T = x @ W1 ----------------
__global__ __launch_bounds__(256) void k_t0(const float* __restrict__ x, const float* __restrict__ W1,
                                            bf16* __restrict__ T, int total) {
    int i = blockIdx.x * blockDim.x + threadIdx.x;
    if (i >= total) return;
    int node = i >> 8, c = i & 255;
    T[i] = f2bf(x[node] * W1[c]);
}

// ---------------- WMMA GEMM: T[NPAD,256] = H[NPAD,256] @ W ----------------
// Block = 256 threads = 8 waves; block tile = 32 rows x 256 cols.
// Wave w: rows = blk*32 + (w>>2)*16, cols = (w&3)*64 -> 4 accumulators.
__global__ __launch_bounds__(256) void k_gemm(const bf16* __restrict__ H, const bf16* __restrict__ Wp,
                                              bf16* __restrict__ T) {
    int wave = threadIdx.x >> 5;
    int lane = threadIdx.x & 31;
    int row0 = blockIdx.x * 32 + (wave >> 2) * 16;
    int col0 = (wave & 3) * 64;
    int m    = lane & 15;
    int half = lane >> 4;

    v8f acc[4] = {v8f{}, v8f{}, v8f{}, v8f{}};

    const bf16* hrow = H + (size_t)(row0 + m) * HID + half * 8;

#pragma unroll
    for (int kb = 0; kb < HID; kb += 32) {
        union { v16bf v; v8bf h[2]; } a;
        a.h[0] = *(const v8bf*)(hrow + kb);        // K = kb + half*8 + 0..7
        a.h[1] = *(const v8bf*)(hrow + kb + 16);   // K = kb + 16 + half*8 + 0..7
#pragma unroll
        for (int t = 0; t < 4; ++t) {
            int fid = (kb >> 5) * 16 + ((col0 >> 4) + t);
            v16bf b = *(const v16bf*)(Wp + ((size_t)fid * 32 + lane) * 16);
            acc[t] = __builtin_amdgcn_wmma_f32_16x16x32_bf16(
                false, a.v, false, b, (short)0, acc[t], false, false);
        }
    }

#pragma unroll
    for (int t = 0; t < 4; ++t) {
        int col = col0 + t * 16 + m;
#pragma unroll
        for (int r = 0; r < 8; ++r) {
            int row = row0 + r + 8 * half;          // C/D layout: VGPR r -> M=r (+8 for hi half)
            T[(size_t)row * HID + col] = f2bf(acc[t][r]);
        }
    }
}

// ---------------- aggregation ----------------
// Self-loop also initializes the accumulator: acc[i] = dinv[i]^2 * T[i]
__global__ __launch_bounds__(256) void k_selfinit(const bf16* __restrict__ T, const float* __restrict__ dinv,
                                                  float* __restrict__ acc, int n) {
    int node = blockIdx.x * (blockDim.x >> 5) + (threadIdx.x >> 5);
    if (node >= n) return;
    int lane = threadIdx.x & 31;
    float w = dinv[node] * dinv[node];
    v8bf v = *(const v8bf*)(T + (size_t)node * HID + lane * 8);
    float* d = acc + (size_t)node * HID + lane * 8;
#pragma unroll
    for (int j = 0; j < 8; ++j) d[j] = w * bf2f(v[j]);
}

// Wave per edge: atomicAdd(acc[dst], dinv[src]*dinv[dst] * T[src])
__global__ __launch_bounds__(256) void k_edges(const int* __restrict__ ei, const bf16* __restrict__ T,
                                               const float* __restrict__ dinv, float* __restrict__ acc, int nE) {
    int e = blockIdx.x * (blockDim.x >> 5) + (threadIdx.x >> 5);
    if (e >= nE) return;
    int lane = threadIdx.x & 31;
    int s = ei[e];
    int d = ei[nE + e];
    float w = dinv[s] * dinv[d];
    v8bf v = *(const v8bf*)(T + (size_t)s * HID + lane * 8);
    float* out = acc + (size_t)d * HID + lane * 8;
#pragma unroll
    for (int j = 0; j < 8; ++j) atomicAdd(out + j, w * bf2f(v[j]));
}

// H = relu(acc + bias) -> bf16
__global__ __launch_bounds__(256) void k_finalize(const float* __restrict__ acc, const float* __restrict__ bias,
                                                  bf16* __restrict__ Hout, int total) {
    int i = blockIdx.x * blockDim.x + threadIdx.x;
    if (i >= total) return;
    float v = acc[i] + bias[i & 255];
    Hout[i] = f2bf(v > 0.0f ? v : 0.0f);
}

// ---------------- readout: out[i] = H[i,:] . Wc + bc ----------------
__global__ __launch_bounds__(256) void k_readout(const bf16* __restrict__ H, const float* __restrict__ Wc,
                                                 const float* __restrict__ bc, float* __restrict__ out, int n) {
    int node = blockIdx.x * (blockDim.x >> 5) + (threadIdx.x >> 5);
    if (node >= n) return;
    int lane = threadIdx.x & 31;
    v8bf v = *(const v8bf*)(H + (size_t)node * HID + lane * 8);
    float s = 0.0f;
#pragma unroll
    for (int j = 0; j < 8; ++j) s += bf2f(v[j]) * Wc[lane * 8 + j];
#pragma unroll
    for (int off = 16; off > 0; off >>= 1) s += __shfl_down(s, off, 32);
    if (lane == 0) out[node] = s + bc[0];
}

// ---------------- host orchestration ----------------
static inline size_t align256(size_t x) { return (x + 255) & ~(size_t)255; }

extern "C" void kernel_launch(void* const* d_in, const int* in_sizes, int n_in,
                              void* d_out, int out_size, void* d_ws, size_t ws_size,
                              hipStream_t stream) {
    const float* x  = (const float*)d_in[0];
    const int*   ei = (const int*)d_in[1];
    const float* W1 = (const float*)d_in[2];
    const float* b1 = (const float*)d_in[3];
    const float* W  = (const float*)d_in[4];
    const float* b  = (const float*)d_in[5];
    const float* Wc = (const float*)d_in[6];
    const float* bc = (const float*)d_in[7];
    float* out = (float*)d_out;

    const int nN = in_sizes[0];          // 50000
    const int nE = in_sizes[1] / 2;      // 800000
    const int NPAD = (nN + 31) & ~31;    // pad rows for boundless WMMA tiles
    const int total = nN * HID;

    char* ws = (char*)d_ws;
    size_t off = 0;
    float* dinv = (float*)(ws + off); off += align256((size_t)nN * 4);
    bf16*  Wp   = (bf16*)(ws + off);  off += align256((size_t)128 * 32 * 16 * 2);
    bf16*  Hb   = (bf16*)(ws + off);  off += align256((size_t)NPAD * HID * 2);
    bf16*  Tb   = (bf16*)(ws + off);  off += align256((size_t)NPAD * HID * 2);
    float* accf = (float*)(ws + off); off += align256((size_t)nN * HID * 4);

    const int TB = 256;
    dim3 blk(TB);
    int gNodes   = (nN + TB - 1) / TB;
    int gEdges   = (nE + TB - 1) / TB;
    int gElems   = (total + TB - 1) / TB;
    int gWaveN   = (nN + (TB / 32) - 1) / (TB / 32);   // wave-per-node
    int gWaveE   = (nE + (TB / 32) - 1) / (TB / 32);   // wave-per-edge
    int gGemm    = NPAD / 32;

    // graph normalization
    k_zero_f<<<gNodes, blk, 0, stream>>>(dinv, nN);
    k_deg   <<<gEdges, blk, 0, stream>>>(ei, dinv, nE);
    k_dinv  <<<gNodes, blk, 0, stream>>>(dinv, nN);

    // pack W into WMMA B-fragment layout (layers 1..9)
    k_packw<<<32, dim3(128), 0, stream>>>(W, Wp);

    // layer 0: T = x @ W1 ; aggregate ; relu
    k_t0      <<<gElems, blk, 0, stream>>>(x, W1, Tb, total);
    k_selfinit<<<gWaveN, blk, 0, stream>>>(Tb, dinv, accf, nN);
    k_edges   <<<gWaveE, blk, 0, stream>>>(ei, Tb, dinv, accf, nE);
    k_finalize<<<gElems, blk, 0, stream>>>(accf, b1, Hb, total);

    // layers 1..9 (shared weights)
    for (int L = 1; L < 10; ++L) {
        k_gemm    <<<gGemm,  blk, 0, stream>>>(Hb, Wp, Tb);
        k_selfinit<<<gWaveN, blk, 0, stream>>>(Tb, dinv, accf, nN);
        k_edges   <<<gWaveE, blk, 0, stream>>>(ei, Tb, dinv, accf, nE);
        k_finalize<<<gElems, blk, 0, stream>>>(accf, b, Hb, total);
    }

    // readout
    k_readout<<<gWaveN, blk, 0, stream>>>(Hb, Wc, bc, out, nN);
}
